// SelfAttention_12103217840380
// MI455X (gfx1250) — matrix operand order repaired and checked by reading, not compiled, not run on hardware
//
#include <hip/hip_runtime.h>

// ---------------------------------------------------------------------------
// Cross-attention:  att = input·w_in + (input*dot_scale)@memᵀ  - mask
//                   out = concat(input, softmax(att)@mem)
// B=16, N=M=1024, D=768.
//
// MI455X reasoning: 51.6 GFLOP total; f32 WMMA (16x16x4) would be ~8x slower
// per instruction than bf16 WMMA (16x16x32) and dominate the ~8.6us HBM floor
// (200MB @ 23.3 TB/s). So: convert once to bf16 staging buffers, run both
// GEMMs on v_wmma_f32_16x16x32_bf16 with f32 accumulators. vT is stored
// transposed so *every* WMMA fragment load is a contiguous per-lane
// global_load_b128 — no gathers, no in-loop LDS transposes.
// ---------------------------------------------------------------------------

#define B_ 16
#define N_ 1024
#define M_ 1024
#define D_ 768

typedef __bf16 bf16;
typedef __attribute__((ext_vector_type(16))) bf16  v16bf;
typedef __attribute__((ext_vector_type(8)))  bf16  v8bf;
typedef __attribute__((ext_vector_type(4)))  bf16  v4bf;
typedef __attribute__((ext_vector_type(8)))  float v8f;
typedef __attribute__((ext_vector_type(4)))  float v4f;

static __device__ __forceinline__ v16bf cat8(v8bf lo, v8bf hi) {
  return __builtin_shufflevector(lo, hi, 0,1,2,3,4,5,6,7,8,9,10,11,12,13,14,15);
}

// ---------------------------------------------------------------------------
// Kernel 1: q = bf16(input * dot_scale)  [B,N,D] row-major
//           out[:, :768] = input  (f32 passthrough half of the concat)
// ---------------------------------------------------------------------------
__global__ void prep_q(const float* __restrict__ inp, const float* __restrict__ dsc,
                       bf16* __restrict__ q, float* __restrict__ out) {
  long i4 = ((long)blockIdx.x * 256 + threadIdx.x) * 4;   // over B*N*D
  int  d   = (int)(i4 % D_);
  long row = i4 / D_;                                      // b*N + n
  v4f x = *(const v4f*)(inp + i4);
  v4f s = *(const v4f*)(dsc + d);
  v4bf qv;
#pragma unroll
  for (int j = 0; j < 4; ++j) qv[j] = (bf16)(x[j] * s[j]);
  *(v4bf*)(q + i4) = qv;
  *(v4f*)(out + row * (2 * D_) + d) = x;                   // first half of concat
}

// ---------------------------------------------------------------------------
// Kernel 2: k  = bf16(memory)            [B,M,D] row-major   (QK^T B-operand)
//           vT = bf16(memory) transposed [B,D,M]             (PV   B-operand)
// 32x32 LDS-tiled transpose for coalesced stores.
// ---------------------------------------------------------------------------
__global__ void prep_kv(const float* __restrict__ mem,
                        bf16* __restrict__ k, bf16* __restrict__ vT) {
  __shared__ bf16 tile[32][36];                            // +4 pad: no bank conflicts
  int bid = blockIdx.x;
  int dt  = bid % (D_ / 32);
  int tmp = bid / (D_ / 32);
  int mt  = tmp % (M_ / 32);
  int b   = tmp / (M_ / 32);
  int m0 = mt * 32, d0 = dt * 32;
  int r  = threadIdx.x >> 3;         // 0..31
  int c4 = (threadIdx.x & 7) * 4;    // 0..28

  v4f x = *(const v4f*)(mem + ((long)(b * M_ + m0 + r) * D_ + d0 + c4));
  v4bf kb;
#pragma unroll
  for (int j = 0; j < 4; ++j) { kb[j] = (bf16)x[j]; tile[r][c4 + j] = kb[j]; }
  *(v4bf*)(k + ((long)(b * M_ + m0 + r) * D_ + d0 + c4)) = kb;
  __syncthreads();
  v4bf o;
#pragma unroll
  for (int j = 0; j < 4; ++j) o[j] = tile[c4 + j][r];      // transpose read
  *(v4bf*)(vT + ((long)(b * D_ + d0 + r) * M_ + m0 + c4)) = o;
}

// ---------------------------------------------------------------------------
// Kernel 3: input_dot[b,n] = input[b,n,:] . w_in   (one wave32 per row)
// ---------------------------------------------------------------------------
__global__ void input_dot_k(const float* __restrict__ inp, const float* __restrict__ w,
                            float* __restrict__ idot) {
  int wave = threadIdx.x >> 5, lane = threadIdx.x & 31;
  long row = (long)blockIdx.x * 8 + wave;                  // < B*N
  const float* p = inp + row * D_;
  float acc = 0.f;
#pragma unroll 4
  for (int i = lane; i < D_; i += 32) acc += p[i] * w[i];
#pragma unroll
  for (int off = 16; off >= 1; off >>= 1) acc += __shfl_xor(acc, off, 32);
  if (lane == 0) idot[row] = acc;
}

// ---------------------------------------------------------------------------
// Kernel 4: scores via WMMA + bias + mask + softmax -> bf16 weights [B,N,M]
// Block = 8 waves over one 16-row n-tile; wave w owns m in [128w, 128w+128).
// C-fragment layout: VGPR v <-> n-row (v + 8*laneHalf), lane%16 <-> m-col.
// ---------------------------------------------------------------------------
__global__ void __launch_bounds__(256, 1)
scores_softmax(const bf16* __restrict__ q, const bf16* __restrict__ k,
               const float* __restrict__ idot, const int* __restrict__ mask,
               bf16* __restrict__ wgt) {
  __shared__ float sred[8][16];
  int bid = blockIdx.x;
  int nt = bid % (N_ / 16), b = bid / (N_ / 16);
  int n_base = nt * 16;
  int wave = threadIdx.x >> 5, lane = threadIdx.x & 31;
  int lane16 = lane & 15, half = lane >> 4;
  int m_base = wave * 128;

  const bf16* qrow = q + (long)(b * N_ + n_base + lane16) * D_;

  v8f zero = {};
  v8f c[8];
#pragma unroll
  for (int t = 0; t < 8; ++t) c[t] = zero;

  for (int kc = 0; kc < D_; kc += 32) {
    __builtin_prefetch(qrow + kc + 32, 0, 1);              // global_prefetch_b8
    // A 16x32 bf16 fragment: lanes 0-15 K{0..15}, lanes 16-31 K{16..31}
    v16bf a  = *(const v16bf*)(qrow + kc + half * 16);
#pragma unroll
    for (int t = 0; t < 8; ++t) {
      // B 32x16 fragment: lane = col m, lane-half selects K 0-15 / 16-31
      const bf16* kp = k + ((long)(b * M_ + m_base + t * 16 + lane16) * D_ + kc + half * 16);
      v16bf bfr = *(const v16bf*)kp;
      c[t] = __builtin_amdgcn_wmma_f32_16x16x32_bf16(false, a, false, bfr,
                                                     (short)0, c[t], false, false);
    }
  }

  // bias + mask, track row max
  float idv[8], rmax[8];
#pragma unroll
  for (int v = 0; v < 8; ++v) {
    idv[v]  = idot[(long)b * N_ + n_base + half * 8 + v];
    rmax[v] = -3.0e38f;
  }
#pragma unroll
  for (int t = 0; t < 8; ++t) {
    float mk = mask[b * M_ + m_base + t * 16 + lane16] ? 0.0f : -1.0e30f;
#pragma unroll
    for (int v = 0; v < 8; ++v) {
      float s = c[t][v] + idv[v] + mk;
      c[t][v] = s;
      rmax[v] = fmaxf(rmax[v], s);
    }
  }
  // reduce across the 16 lanes of each half (xor 1,2,4,8 stays in-half)
#pragma unroll
  for (int v = 0; v < 8; ++v)
#pragma unroll
    for (int off = 8; off >= 1; off >>= 1)
      rmax[v] = fmaxf(rmax[v], __shfl_xor(rmax[v], off, 32));
  // combine across the 8 waves (different m ranges)
  if (lane16 == 0)
#pragma unroll
    for (int v = 0; v < 8; ++v) sred[wave][half * 8 + v] = rmax[v];
  __syncthreads();
#pragma unroll
  for (int v = 0; v < 8; ++v) {
    float g = -3.0e38f;
    for (int w = 0; w < 8; ++w) g = fmaxf(g, sred[w][half * 8 + v]);
    rmax[v] = g;
  }
  __syncthreads();

  float rsum[8];
#pragma unroll
  for (int v = 0; v < 8; ++v) rsum[v] = 0.f;
#pragma unroll
  for (int t = 0; t < 8; ++t)
#pragma unroll
    for (int v = 0; v < 8; ++v) {
      float p = __expf(c[t][v] - rmax[v]);
      c[t][v] = p;
      rsum[v] += p;
    }
#pragma unroll
  for (int v = 0; v < 8; ++v)
#pragma unroll
    for (int off = 8; off >= 1; off >>= 1)
      rsum[v] += __shfl_xor(rsum[v], off, 32);
  if (lane16 == 0)
#pragma unroll
    for (int v = 0; v < 8; ++v) sred[wave][half * 8 + v] = rsum[v];
  __syncthreads();
#pragma unroll
  for (int v = 0; v < 8; ++v) {
    float g = 0.f;
    for (int w = 0; w < 8; ++w) g += sred[w][half * 8 + v];
    rsum[v] = 1.0f / g;
  }

  // normalized weights -> bf16 [B,N,M] row-major (A-operand of PV GEMM)
#pragma unroll
  for (int v = 0; v < 8; ++v) {
    long o = (long)(b * N_ + n_base + half * 8 + v) * M_ + m_base + lane16;
#pragma unroll
    for (int t = 0; t < 8; ++t) wgt[o + t * 16] = (bf16)(c[t][v] * rsum[v]);
  }
}

// ---------------------------------------------------------------------------
// Kernel 5: out[:, 768:] = weights @ memory  via WMMA over vT [B,D,M].
// Block = 8 waves over one 16-row n-tile; wave w owns d in [96w, 96w+96).
// ---------------------------------------------------------------------------
__global__ void __launch_bounds__(256, 1)
attn_out(const bf16* __restrict__ wgt, const bf16* __restrict__ vT,
         float* __restrict__ out) {
  int bid = blockIdx.x;
  int nt = bid % (N_ / 16), b = bid / (N_ / 16);
  int n_base = nt * 16;
  int wave = threadIdx.x >> 5, lane = threadIdx.x & 31;
  int lane16 = lane & 15, half = lane >> 4;
  int d0 = wave * 96;

  const bf16* wrow = wgt + (long)(b * N_ + n_base + lane16) * M_;

  v8f zero = {};
  v8f c[6];
#pragma unroll
  for (int t = 0; t < 6; ++t) c[t] = zero;

  for (int mc = 0; mc < M_; mc += 32) {
    __builtin_prefetch(wrow + mc + 32, 0, 1);
    v16bf a  = *(const v16bf*)(wrow + mc + half * 16);
#pragma unroll
    for (int t = 0; t < 6; ++t) {
      // B fragment: lane = col d, contiguous m run from transposed V
      const bf16* vp = vT + ((long)(b * D_ + d0 + t * 16 + lane16) * M_ + mc + half * 16);
      v16bf bfr = *(const v16bf*)vp;
      c[t] = __builtin_amdgcn_wmma_f32_16x16x32_bf16(false, a, false, bfr,
                                                     (short)0, c[t], false, false);
    }
  }
#pragma unroll
  for (int t = 0; t < 6; ++t)
#pragma unroll
    for (int v = 0; v < 8; ++v) {
      int n = n_base + half * 8 + v;
      out[(long)(b * N_ + n) * (2 * D_) + D_ + d0 + t * 16 + lane16] = c[t][v];
    }
}

// ---------------------------------------------------------------------------
// Workspace layout (bytes):
//   q   [B,N,D] bf16 : 25,165,824  @ 0
//   k   [B,M,D] bf16 : 25,165,824  @ 25,165,824
//   vT  [B,D,M] bf16 : 25,165,824  @ 50,331,648
//   wgt [B,N,M] bf16 : 33,554,432  @ 75,497,472
//   idot[B,N]   f32  :     65,536  @ 109,051,904     (total ~104 MB)
// ---------------------------------------------------------------------------
extern "C" void kernel_launch(void* const* d_in, const int* in_sizes, int n_in,
                              void* d_out, int out_size, void* d_ws, size_t ws_size,
                              hipStream_t stream) {
  const float* inp  = (const float*)d_in[0];
  const float* mem  = (const float*)d_in[1];
  const int*   mask = (const int*)d_in[2];
  const float* w_in = (const float*)d_in[3];
  const float* dsc  = (const float*)d_in[4];
  float* out = (float*)d_out;

  char* ws = (char*)d_ws;
  bf16*  q    = (bf16*)(ws);
  bf16*  k    = (bf16*)(ws + 25165824L);
  bf16*  vT   = (bf16*)(ws + 50331648L);
  bf16*  wgt  = (bf16*)(ws + 75497472L);
  float* idot = (float*)(ws + 109051904L);

  prep_q      <<<(B_ * N_ * D_) / (4 * 256), 256, 0, stream>>>(inp, dsc, q, out);
  prep_kv     <<<B_ * (M_ / 32) * (D_ / 32), 256, 0, stream>>>(mem, k, vT);
  input_dot_k <<<(B_ * N_) / 8,              256, 0, stream>>>(inp, w_in, idot);
  scores_softmax<<<B_ * (N_ / 16),           256, 0, stream>>>(q, k, idot, mask, wgt);
  attn_out    <<<B_ * (N_ / 16),             256, 0, stream>>>(wgt, vT, out);
}
